// TemporalMambaFusionC1_89945205113400
// MI455X (gfx1250) — matrix operand order repaired and checked by reading, not compiled, not run on hardware
//
#include <hip/hip_runtime.h>
#include <hip/hip_bf16.h>
#include <math.h>

// ---------------------------------------------------------------------------
// TemporalMambaFusion for MI455X (gfx1250, wave32, WMMA).
//
// Layout decisions:
//  * activations for the SSM part: f16 row-major [M=262144][128]  (M = n*8+t)
//  * all GEMMs via v_wmma_f32_16x16x32_f16, f32 accumulate
//  * entire SSM block fused in ONE kernel; recurrence runs inside the WMMA
//    accumulator layout (lane's 8 VGPRs == t=0..7 of one sequence)
//  * convs as implicit GEMM on NHWC f16 with pre-reordered f16 weights
//
// Param pointer mapping assumes jax-style flatten of the params dict
// (dict keys sorted alphabetically, lists in order) after the two tensors:
//  0 temporal_features, 1 sup,
//  2..11  block0: ffn_b1,ffn_b2,ffn_w1,ffn_w2,gate_b,gate_w,in_w,out_w,sup_b,sup_w
//  12..21 block1 (same order)
//  22 post_conv_w, 23 post_gn_b, 24 post_gn_s, 25 rc1_w, 26 rc2_w,
//  27 rg1_b, 28 rg1_s, 29 rg2_b, 30 rg2_s
//
// Needs ws_size >= ~187 MB.
// ---------------------------------------------------------------------------

typedef _Float16 h16;
typedef h16  v16h __attribute__((ext_vector_type(16)));
typedef h16  v8h  __attribute__((ext_vector_type(8)));
typedef float v8f __attribute__((ext_vector_type(8)));

#define CDIM   128
#define HID    256
#define TDIM   8
#define HWSZ   16384            // 128*128
#define NSEQ   32768            // B*H*W
#define MROWS  262144           // NSEQ*T
#define LDX    136              // padded f16 LDS stride for K=128 (68 DW, +4 bank rot)
#define LDH2   264              // padded f16 LDS stride for K=256 (132 DW, +4 bank rot)
#define LDCV   136              // conv LDS channel stride

__device__ __forceinline__ float geluf(float x) {
    return 0.5f * x * (1.0f + erff(x * 0.70710678118654752f));
}
__device__ __forceinline__ float sigmf(float x) {
    return 1.0f / (1.0f + expf(-x));
}

// ---- WMMA fragment loaders (row-major f16 sources) -------------------------
// A (16x32, MxK): low lanes hold K{k0..k0+7, k0+16..k0+23}, high lanes +8.
__device__ __forceinline__ v16h a_frag(const h16* base, int ld, int row0, int k0, int lane) {
    const int m  = lane & 15;
    const int ko = (lane & 16) ? 8 : 0;
    const h16* p = base + (size_t)(row0 + m) * ld + k0 + ko;
    v8h lo = *(const v8h*)p;
    v8h hi = *(const v8h*)(p + 16);
    v16h r;
#pragma unroll
    for (int i = 0; i < 8; ++i) { r[i] = lo[i]; r[i + 8] = hi[i]; }
    return r;
}
// B (32x16, KxN) from W stored row-major [N][K] (i.e. x @ W^T):
// low lanes hold K{k0..k0+7, k0+16..k0+23} of column n, high lanes +8.
__device__ __forceinline__ v16h b_frag(const h16* base, int ld, int col0, int k0, int lane) {
    const int n  = lane & 15;
    const int ko = (lane & 16) ? 8 : 0;
    const h16* p = base + (size_t)(col0 + n) * ld + k0 + ko;
    v8h lo = *(const v8h*)p;
    v8h hi = *(const v8h*)(p + 16);
    v16h r;
#pragma unroll
    for (int i = 0; i < 8; ++i) { r[i] = lo[i]; r[i + 8] = hi[i]; }
    return r;
}

template <int NT>
__device__ __forceinline__ void zero_acc(v8f* acc) {
#pragma unroll
    for (int nt = 0; nt < NT; ++nt)
#pragma unroll
        for (int j = 0; j < 8; ++j) acc[nt][j] = 0.0f;
}

// acc[NT] += A(rows rowBase..+16 of As, KxLDA row-major) x W^T (NT*16 cols)
template <int NT, int K, int LDA>
__device__ __forceinline__ void gemm_acc(const h16* As, int rowBase,
                                         const h16* Bw, int ldb,
                                         v8f* acc, int lane) {
#pragma unroll
    for (int k0 = 0; k0 < K; k0 += 32) {
        v16h a = a_frag(As, LDA, rowBase, k0, lane);
#pragma unroll
        for (int nt = 0; nt < NT; ++nt) {
            v16h b = b_frag(Bw, ldb, nt * 16, k0, lane);
            acc[nt] = __builtin_amdgcn_wmma_f32_16x16x32_f16(
                false, a, false, b, (short)0, acc[nt], false, false);
        }
    }
}

// ---------------------------------------------------------------------------
// K0: bilinear upsample of sup (32->128, half-pixel), cloud mask, and
// transpose [B,T,C,H,W] f32 -> x16 [n*8+t][c] f16 (LDS-staged, coalesced).
// grid: B*H blocks of 256 threads; dyn-LDS = 70144 B.
// ---------------------------------------------------------------------------
__global__ void prep_kernel(const float* __restrict__ tf, const float* __restrict__ sup,
                            h16* __restrict__ x16, float* __restrict__ supf) {
    extern __shared__ char smem[];
    float* sl = (float*)smem;                 // [8][128] sup row
    h16*   xt = (h16*)(smem + 8 * 128 * 4);   // [32][1032]
    const int b = blockIdx.x >> 7, h = blockIdx.x & 127;
    const int tid = threadIdx.x;

    for (int i = tid; i < 1024; i += 256) {
        int t = i >> 7, w = i & 127;
        float sy = h * 0.25f - 0.375f, sx = w * 0.25f - 0.375f;
        int y0 = (int)floorf(sy), x0 = (int)floorf(sx);
        float fy = sy - (float)y0, fx = sx - (float)x0;
        int y0c = min(max(y0, 0), 31), y1c = min(max(y0 + 1, 0), 31);
        int x0c = min(max(x0, 0), 31), x1c = min(max(x0 + 1, 0), 31);
        const float* sp = sup + (size_t)(b * 8 + t) * 1024;
        float v = sp[y0c * 32 + x0c] * (1 - fy) * (1 - fx) + sp[y0c * 32 + x1c] * (1 - fy) * fx +
                  sp[y1c * 32 + x0c] * fy * (1 - fx)       + sp[y1c * 32 + x1c] * fy * fx;
        sl[i] = v;
        supf[(size_t)(b * HWSZ + h * 128 + w) * 8 + t] = v;
    }
    __syncthreads();
    for (int wc = 0; wc < 128; wc += 32) {
        for (int i = tid; i < 32 * 1024; i += 256) {     // coalesced reads along w
            int wl = i & 31, tc = i >> 5;
            int t = tc >> 7, c = tc & 127, w = wc + wl;
            float v = tf[(((size_t)(b * 8 + t) * 128 + c) * 128 + h) * 128 + w] *
                      (1.0f - sl[t * 128 + w]);
            xt[wl * 1032 + tc] = (h16)v;
        }
        __syncthreads();
        for (int i = tid; i < 32 * 128; i += 256) {      // coalesced 16B writes
            int wl = i >> 7, k8 = (i & 127) * 8;
            size_t n = (size_t)b * HWSZ + h * 128 + wc + wl;
            *(v8h*)(x16 + n * 1024 + k8) = *(const v8h*)(xt + wl * 1032 + k8);
        }
        __syncthreads();
    }
}

// ---------------------------------------------------------------------------
// Fused SSM block: one workgroup = 128 rows = 16 sequences x 8 timesteps.
// Each wave owns 16 rows -> fully wave-local (no barriers).
// grid: 2048 x 256 threads; dyn-LDS = 137216 B.
// ---------------------------------------------------------------------------
__launch_bounds__(256, 1)
__global__ void ssm_block_kernel(const h16* __restrict__ xg, const float* __restrict__ supf,
                                 const h16* __restrict__ w_in, const h16* __restrict__ w_f1,
                                 const h16* __restrict__ w_f2, const h16* __restrict__ w_g,
                                 const h16* __restrict__ w_o,
                                 const float* __restrict__ b_f1, const float* __restrict__ b_f2,
                                 const float* __restrict__ b_g, const float* __restrict__ sup_w,
                                 const float* __restrict__ sup_b,
                                 h16* __restrict__ xout, h16* __restrict__ pool, int do_mean) {
    extern __shared__ char smem[];
    h16* xs   = (h16*)smem;          // [128][136] x -> xt -> s_seq
    h16* xins = xs + 128 * LDX;      // [128][136] xin
    h16* hs   = xins + 128 * LDX;    // [128][264] ffn hidden

    const int  tid = threadIdx.x, lane = tid & 31, wv = tid >> 5;
    const int  rowBase = wv * 16;
    const long tileRow = (long)blockIdx.x * 128;
    const int  m  = lane & 15;
    const int  hi = (lane & 16) ? 8 : 0;

    __builtin_prefetch(w_in, 0, 1);
    __builtin_prefetch(w_f1, 0, 1);
    __builtin_prefetch(w_f2, 0, 1);

    // wave-local load of this wave's 16 rows into LDS
    for (int i = lane; i < 16 * 16; i += 32) {
        int r = rowBase + (i >> 4), c8 = (i & 15) * 8;
        *(v8h*)(xs + r * LDX + c8) = *(const v8h*)(xg + (tileRow + r) * CDIM + c8);
    }

    // ---- xin = x @ in_w^T -------------------------------------------------
    {
        v8f acc[8]; zero_acc<8>(acc);
        gemm_acc<8, 128, LDX>(xs, rowBase, w_in, 128, acc, lane);
#pragma unroll
        for (int nt = 0; nt < 8; ++nt) { int col = nt * 16 + m;
#pragma unroll
            for (int v = 0; v < 8; ++v)
                xins[(rowBase + hi + v) * LDX + col] = (h16)acc[nt][v];
        }
    }
    // ---- hmid = gelu(xin @ w1^T + b1) ------------------------------------
    {
        v8f acc[16]; zero_acc<16>(acc);
        gemm_acc<16, 128, LDX>(xins, rowBase, w_f1, 128, acc, lane);
#pragma unroll
        for (int nt = 0; nt < 16; ++nt) { int col = nt * 16 + m; float bb = b_f1[col];
#pragma unroll
            for (int v = 0; v < 8; ++v)
                hs[(rowBase + hi + v) * LDH2 + col] = (h16)geluf(acc[nt][v] + bb);
        }
    }
    // ---- xt = xin + hmid @ w2^T + b2  (kept in regs AND stored to LDS) ----
    v8f xt[8]; zero_acc<8>(xt);
    {
        gemm_acc<8, 256, LDH2>(hs, rowBase, w_f2, 256, xt, lane);
#pragma unroll
        for (int nt = 0; nt < 8; ++nt) { int col = nt * 16 + m; float bb = b_f2[col];
#pragma unroll
            for (int v = 0; v < 8; ++v) {
                float t = xt[nt][v] + bb + (float)xins[(rowBase + hi + v) * LDX + col];
                xt[nt][v] = t;
                xs[(rowBase + hi + v) * LDX + col] = (h16)t;
            }
        }
    }
    // ---- gate + fused recurrence (lane VGPR v == timestep t) -------------
    const long grow0 = tileRow + rowBase + hi;   // multiple of 8: seq*8 + t=0
    float supv[8];
#pragma unroll
    for (int v = 0; v < 8; ++v) supv[v] = supf[grow0 + v];
    {
        v8f g[8]; zero_acc<8>(g);
        gemm_acc<8, 128, LDX>(xs, rowBase, w_g, 128, g, lane);
#pragma unroll
        for (int nt = 0; nt < 8; ++nt) { int col = nt * 16 + m;
            float gb = b_g[col], sw = sup_w[col], sb = sup_b[col];
            float s = 0.0f;
#pragma unroll
            for (int v = 0; v < 8; ++v) {
                float gg = sigmf(g[nt][v] + gb) * sigmf(supv[v] * sw + sb);
                s = (1.0f - gg) * s + gg * xt[nt][v];
                xs[(rowBase + hi + v) * LDX + col] = (h16)s;
            }
        }
    }
    // ---- out = s_seq @ out_w^T -------------------------------------------
    {
        v8f o[8]; zero_acc<8>(o);
        gemm_acc<8, 128, LDX>(xs, rowBase, w_o, 128, o, lane);
        if (do_mean) {
            long seq = grow0 >> 3;
#pragma unroll
            for (int nt = 0; nt < 8; ++nt) { int col = nt * 16 + m;
                float s = 0.0f;
#pragma unroll
                for (int v = 0; v < 8; ++v) s += o[nt][v];
                pool[seq * CDIM + col] = (h16)(s * 0.125f);
            }
        } else {
#pragma unroll
            for (int nt = 0; nt < 8; ++nt) { int col = nt * 16 + m;
#pragma unroll
                for (int v = 0; v < 8; ++v)
                    xout[(grow0 + v) * CDIM + col] = (h16)o[nt][v];
            }
        }
    }
}

// ---------------------------------------------------------------------------
// Conv3x3 (SAME) as implicit WMMA GEMM on NHWC f16.
// grid: B*H blocks; wave w handles 16 output pixels; dyn-LDS = 106080 B.
// Weights pre-reordered to f16 [tap][cout][cin].
// ---------------------------------------------------------------------------
__launch_bounds__(256, 1)
__global__ void conv3_kernel(const h16* __restrict__ in16, const h16* __restrict__ wtap,
                             float* __restrict__ out) {
    extern __shared__ char smem[];
    h16* tile = (h16*)smem;   // [3][130][136]
    const int b = blockIdx.x >> 7, h = blockIdx.x & 127;
    const int tid = threadIdx.x, lane = tid & 31, wv = tid >> 5;

    for (int i = tid; i < 3 * 130 * 16; i += 256) {
        int c8 = (i & 15) * 8;
        int wp = (i >> 4) % 130;
        int r3 = (i >> 4) / 130;
        int hsrc = h + r3 - 1, wsrc = wp - 1;
        v8h val;
#pragma unroll
        for (int j = 0; j < 8; ++j) val[j] = (h16)0.0f;
        if (hsrc >= 0 && hsrc < 128 && wsrc >= 0 && wsrc < 128)
            val = *(const v8h*)(in16 + (size_t)(b * HWSZ + hsrc * 128 + wsrc) * CDIM + c8);
        *(v8h*)(tile + (r3 * 130 + wp) * LDCV + c8) = val;
    }
    __syncthreads();

    const int w0 = wv * 16;
    v8f acc[8]; zero_acc<8>(acc);
#pragma unroll
    for (int tap = 0; tap < 9; ++tap) {
        const int dy = tap / 3, dx = tap % 3;
        const h16* arow = tile + (dy * 130 + w0 + dx) * LDCV;   // A row m = pixel w0+m
        const h16* brow = wtap + (size_t)tap * CDIM * CDIM;
        gemm_acc<8, 128, LDCV>(arow, 0, brow, 128, acc, lane);
    }
    const int m = lane & 15, hi = (lane & 16) ? 8 : 0;
#pragma unroll
    for (int nt = 0; nt < 8; ++nt) { int col = nt * 16 + m;
#pragma unroll
        for (int v = 0; v < 8; ++v)
            out[(size_t)(b * HWSZ + h * 128 + w0 + hi + v) * CDIM + col] = acc[nt][v];
    }
}

// ---------------------------------------------------------------------------
// GroupNorm stats (16 groups of 8 ch): grid 32 = (b,g); population var.
// ---------------------------------------------------------------------------
__global__ void gn_stats_kernel(const float* __restrict__ x, float2* __restrict__ stats) {
    const int bg = blockIdx.x, b = bg >> 4, g = bg & 15;
    const int tid = threadIdx.x;
    const float* base = x + (size_t)b * HWSZ * CDIM + g * 8;
    float s1 = 0.0f, s2 = 0.0f;
    for (int pix = tid; pix < HWSZ; pix += 256) {
        const float* p = base + (size_t)pix * CDIM;
#pragma unroll
        for (int c = 0; c < 8; ++c) { float v = p[c]; s1 += v; s2 += v * v; }
    }
    __shared__ float r1[256], r2[256];
    r1[tid] = s1; r2[tid] = s2;
    __syncthreads();
    for (int s = 128; s > 0; s >>= 1) {
        if (tid < s) { r1[tid] += r1[tid + s]; r2[tid] += r2[tid + s]; }
        __syncthreads();
    }
    if (tid == 0) {
        float mean = r1[0] * (1.0f / 131072.0f);
        float var  = r2[0] * (1.0f / 131072.0f) - mean * mean;
        stats[bg] = make_float2(mean, rsqrtf(var + 1e-5f));
    }
}

// GN apply + GELU; writes f16 (next conv input) and/or f32 copies.
__global__ void gn_apply_kernel(const float* __restrict__ x, const float2* __restrict__ st,
                                const float* __restrict__ sc, const float* __restrict__ bi,
                                h16* __restrict__ o16, float* __restrict__ o32, int total) {
    int idx = blockIdx.x * 256 + threadIdx.x;
    if (idx >= total) return;
    int c = idx & 127;
    int b = (idx >> 7) >= HWSZ ? 1 : 0;
    float2 s = st[b * 16 + (c >> 3)];
    float v = geluf((x[idx] - s.x) * s.y * sc[c] + bi[c]);
    if (o16) o16[idx] = (h16)v;
    if (o32) o32[idx] = v;
}

// Final stage: GN apply + GELU + residual add + NHWC->NCHW transpose.
__global__ void gn_final_kernel(const float* __restrict__ x, const float2* __restrict__ st,
                                const float* __restrict__ sc, const float* __restrict__ bi,
                                const float* __restrict__ yres, float* __restrict__ out, int total) {
    int idx = blockIdx.x * 256 + threadIdx.x;   // NCHW order -> coalesced writes
    if (idx >= total) return;
    int w = idx & 127, hh = (idx >> 7) & 127, c = (idx >> 14) & 127, b = idx >> 21;
    size_t src = (size_t)(b * HWSZ + hh * 128 + w) * CDIM + c;
    float2 s = st[b * 16 + (c >> 3)];
    float v = geluf((x[src] - s.x) * s.y * sc[c] + bi[c]);
    out[idx] = yres[src] + v;
}

// ---- weight prep ----------------------------------------------------------
__global__ void cvt_f16_kernel(const float* __restrict__ in, h16* __restrict__ out, int n) {
    int i = blockIdx.x * 256 + threadIdx.x;
    if (i < n) out[i] = (h16)in[i];
}
// OIHW (128,128,3,3) f32 -> [tap][o][i] f16
__global__ void conv_w_prep_kernel(const float* __restrict__ w, h16* __restrict__ out) {
    int i = blockIdx.x * 256 + threadIdx.x;
    if (i >= CDIM * CDIM * 9) return;
    int kk = i % 9, ci = (i / 9) % CDIM, co = i / (9 * CDIM);
    out[((size_t)kk * CDIM + co) * CDIM + ci] = (h16)w[i];
}

// ---------------------------------------------------------------------------
extern "C" void kernel_launch(void* const* d_in, const int* in_sizes, int n_in,
                              void* d_out, int out_size, void* d_ws, size_t ws_size,
                              hipStream_t stream) {
    (void)in_sizes; (void)n_in; (void)out_size; (void)ws_size;
    const float* tf  = (const float*)d_in[0];
    const float* sup = (const float*)d_in[1];
    // per-block params (sorted-key order within each block dict)
    const float *ffn_b1[2], *ffn_b2[2], *ffn_w1[2], *ffn_w2[2], *gate_b[2],
                *gate_w[2], *in_w[2], *out_w[2], *sup_b[2], *sup_w[2];
    for (int i = 0; i < 2; ++i) {
        void* const* p = d_in + 2 + 10 * i;
        ffn_b1[i] = (const float*)p[0]; ffn_b2[i] = (const float*)p[1];
        ffn_w1[i] = (const float*)p[2]; ffn_w2[i] = (const float*)p[3];
        gate_b[i] = (const float*)p[4]; gate_w[i] = (const float*)p[5];
        in_w[i]   = (const float*)p[6]; out_w[i]  = (const float*)p[7];
        sup_b[i]  = (const float*)p[8]; sup_w[i]  = (const float*)p[9];
    }
    const float* post_conv_w = (const float*)d_in[22];
    const float* post_gn_b   = (const float*)d_in[23];
    const float* post_gn_s   = (const float*)d_in[24];
    const float* rc1_w       = (const float*)d_in[25];
    const float* rc2_w       = (const float*)d_in[26];
    const float* rg1_b       = (const float*)d_in[27];
    const float* rg1_s       = (const float*)d_in[28];
    const float* rg2_b       = (const float*)d_in[29];
    const float* rg2_s       = (const float*)d_in[30];

    // ---- workspace carve (~187 MB) ----
    char* cur = (char*)d_ws;
    auto take = [&](size_t bytes) -> char* {
        char* r = cur; cur += (bytes + 255) & ~(size_t)255; return r;
    };
    h16*   X16A  = (h16*)take((size_t)MROWS * CDIM * 2);   // 64 MB
    h16*   X16B  = (h16*)take((size_t)MROWS * CDIM * 2);   // 64 MB
    float* SUPF  = (float*)take((size_t)MROWS * 4);        // 1 MB
    float* CONVF = (float*)take((size_t)NSEQ * CDIM * 4);  // 16.8 MB
    float* YF32  = (float*)take((size_t)NSEQ * CDIM * 4);  // 16.8 MB
    h16*   P16A  = (h16*)take((size_t)NSEQ * CDIM * 2);    // 8.4 MB
    h16*   P16B  = (h16*)take((size_t)NSEQ * CDIM * 2);    // 8.4 MB
    h16 *W_IN[2], *W_G[2], *W_O[2], *W_F1[2], *W_F2[2];
    for (int i = 0; i < 2; ++i) {
        W_IN[i] = (h16*)take(16384 * 2); W_G[i] = (h16*)take(16384 * 2);
        W_O[i]  = (h16*)take(16384 * 2);
        W_F1[i] = (h16*)take(32768 * 2); W_F2[i] = (h16*)take(32768 * 2);
    }
    h16* WT1 = (h16*)take((size_t)9 * CDIM * CDIM * 2);
    h16* WT2 = (h16*)take((size_t)9 * CDIM * CDIM * 2);
    h16* WT3 = (h16*)take((size_t)9 * CDIM * CDIM * 2);
    float2* STATS = (float2*)take(32 * sizeof(float2));

    const int ssmShmem  = 128 * LDX * 2 * 2 + 128 * LDH2 * 2;   // 137216
    const int convShmem = 3 * 130 * LDCV * 2;                   // 106080
    const int prepShmem = 8 * 128 * 4 + 32 * 1032 * 2;          // 70144
    hipFuncSetAttribute((const void*)ssm_block_kernel,
                        hipFuncAttributeMaxDynamicSharedMemorySize, ssmShmem);
    hipFuncSetAttribute((const void*)conv3_kernel,
                        hipFuncAttributeMaxDynamicSharedMemorySize, convShmem);
    hipFuncSetAttribute((const void*)prep_kernel,
                        hipFuncAttributeMaxDynamicSharedMemorySize, prepShmem);

    // ---- weight prep ----
    for (int i = 0; i < 2; ++i) {
        cvt_f16_kernel<<<64, 256, 0, stream>>>(in_w[i],   W_IN[i], 16384);
        cvt_f16_kernel<<<64, 256, 0, stream>>>(gate_w[i], W_G[i],  16384);
        cvt_f16_kernel<<<64, 256, 0, stream>>>(out_w[i],  W_O[i],  16384);
        cvt_f16_kernel<<<128, 256, 0, stream>>>(ffn_w1[i], W_F1[i], 32768);
        cvt_f16_kernel<<<128, 256, 0, stream>>>(ffn_w2[i], W_F2[i], 32768);
    }
    conv_w_prep_kernel<<<576, 256, 0, stream>>>(post_conv_w, WT1);
    conv_w_prep_kernel<<<576, 256, 0, stream>>>(rc1_w, WT2);
    conv_w_prep_kernel<<<576, 256, 0, stream>>>(rc2_w, WT3);

    // ---- input prep: mask + transpose ----
    prep_kernel<<<256, 256, prepShmem, stream>>>(tf, sup, X16A, SUPF);

    // ---- SSM blocks ----
    ssm_block_kernel<<<2048, 256, ssmShmem, stream>>>(
        X16A, SUPF, W_IN[0], W_F1[0], W_F2[0], W_G[0], W_O[0],
        ffn_b1[0], ffn_b2[0], gate_b[0], sup_w[0], sup_b[0],
        X16B, (h16*)nullptr, 0);
    ssm_block_kernel<<<2048, 256, ssmShmem, stream>>>(
        X16B, SUPF, W_IN[1], W_F1[1], W_F2[1], W_G[1], W_O[1],
        ffn_b1[1], ffn_b2[1], gate_b[1], sup_w[1], sup_b[1],
        (h16*)nullptr, P16A, 1);   // fused temporal mean -> NHWC f16

    const int totNC = NSEQ * CDIM;          // 4194304
    const int gridNC = (totNC + 255) / 256;

    // ---- y = conv_gn_act(pool) ----
    conv3_kernel<<<256, 256, convShmem, stream>>>(P16A, WT1, CONVF);
    gn_stats_kernel<<<32, 256, 0, stream>>>(CONVF, STATS);
    gn_apply_kernel<<<gridNC, 256, 0, stream>>>(CONVF, STATS, post_gn_s, post_gn_b,
                                                P16B, YF32, totNC);
    // ---- r = conv_gn_act(y) ----
    conv3_kernel<<<256, 256, convShmem, stream>>>(P16B, WT2, CONVF);
    gn_stats_kernel<<<32, 256, 0, stream>>>(CONVF, STATS);
    gn_apply_kernel<<<gridNC, 256, 0, stream>>>(CONVF, STATS, rg1_s, rg1_b,
                                                P16A, (float*)nullptr, totNC);
    // ---- r = conv_gn_act(r); out = y + r (fused, NCHW) ----
    conv3_kernel<<<256, 256, convShmem, stream>>>(P16A, WT3, CONVF);
    gn_stats_kernel<<<32, 256, 0, stream>>>(CONVF, STATS);
    gn_final_kernel<<<gridNC, 256, 0, stream>>>(CONVF, STATS, rg2_s, rg2_b,
                                                YF32, (float*)d_out, totNC);
}
